// NetVLAD_635655160323
// MI455X (gfx1250) — compile-verified, matrix operand order checked by reading, not run on hardware
//
#include <hip/hip_runtime.h>
#include <math.h>

#define NN 32
#define CC 128
#define KK 64
#define SS 4096
#define TS 128          // pixels per block
#define XNS 136         // xn row stride (halves): 272B, 16B-aligned rows
#define XTS 136         // xnT row stride (halves)
#define CAS 136         // cen/a row stride (halves)
#define LGS 132         // logits row stride (floats)
#define ALPHA 100.0f
#define EPSN 1e-12f

typedef __attribute__((ext_vector_type(16))) _Float16 v16h;
typedef __attribute__((ext_vector_type(8)))  _Float16 v8h;
typedef __attribute__((ext_vector_type(8)))  float    v8f;

__device__ __forceinline__ v16h frag2(const _Float16* base0, const _Float16* base1) {
  v8h a = *(const v8h*)base0;   // 16B aligned -> ds_load_b128
  v8h b = *(const v8h*)base1;
  return __builtin_shufflevector(a, b, 0,1,2,3,4,5,6,7,8,9,10,11,12,13,14,15);
}

__global__ void zero_ws_kernel(float* ws, int n) {
  int i = blockIdx.x * 256 + threadIdx.x;
  if (i < n) ws[i] = 0.0f;
}

// Fused: per-pixel L2 norm -> logits GEMM (WMMA) -> softmax -> aggregation GEMM (WMMA)
__global__ __launch_bounds__(256)
void netvlad_main(const float* __restrict__ x, const float* __restrict__ cent,
                  float* __restrict__ vlad, float* __restrict__ asum) {
  __shared__ __align__(16) _Float16 xn [CC * XNS];  // x-hat, [c][s]  (phase-4 B)
  __shared__ __align__(16) _Float16 xnT[TS * XTS];  // x-hat, [s][c]  (phase-2 B)
  __shared__ __align__(16) _Float16 ca [KK * CAS];  // centroids f16, then assignments f16
  __shared__ float lg[KK * LGS];                    // logits then softmax probs (f32)
  __shared__ float biasl[KK];
  __shared__ float ssp[2 * TS];
  __shared__ float rns[TS];

  const int tid  = threadIdx.x;
  const int lane = tid & 31;
  const int wid  = tid >> 5;           // 8 waves
  const int rlo  = lane & 15;
  const int hi   = lane >> 4;          // 0 or 1

  const int n  = blockIdx.x >> 5;      // SS/TS == 32 tiles per image
  const int s0 = (blockIdx.x & 31) * TS;

  // ---- Phase 0: centroids -> f16 LDS; bias[k] = -alpha * ||centroid_k|| ----
  for (int idx = tid; idx < KK * CC; idx += 256) {
    int k = idx >> 7, c = idx & 127;
    ca[k * CAS + c] = (_Float16)cent[idx];
  }
  if (tid < KK) {
    float ss = 0.f;
    const float* cr = cent + tid * CC;
    for (int c = 0; c < CC; ++c) { float v = cr[c]; ss += v * v; }
    biasl[tid] = -ALPHA * sqrtf(ss);
  }

  // ---- Phase 1: load x tile (coalesced over s), per-pixel sumsq, normalize ----
  {
    const int p = tid & (TS - 1);
    const int g = tid >> 7;            // channel half: [g*64, g*64+64)
    const float* xp = x + ((long)n * CC + g * 64) * SS + s0 + p;
    float ss = 0.f;
    for (int cc = 0; cc < 64; ++cc) {
      float v = xp[(long)cc * SS];
      ss += v * v;
      xn[(g * 64 + cc) * XNS + p] = (_Float16)v;
    }
    ssp[g * TS + p] = ss;
  }
  __syncthreads();
  if (tid < TS) {
    float tot = ssp[tid] + ssp[TS + tid];
    rns[tid] = 1.0f / fmaxf(sqrtf(tot), EPSN);
  }
  __syncthreads();
  {
    const int p = tid & (TS - 1);
    const int g = tid >> 7;
    float rn = rns[p];
    for (int cc = 0; cc < 64; ++cc) {
      int c = g * 64 + cc;
      int i = c * XNS + p;
      _Float16 hv = (_Float16)((float)xn[i] * rn);
      xn[i] = hv;                      // [c][s] layout
      xnT[p * XTS + c] = hv;           // [s][c] layout
    }
  }
  __syncthreads();

  // ---- Phase 2: logits[64 x TS] = 2a * cent[64x128] @ xhat[128 x TS] + bias ----
  for (int i = 0; i < 4; ++i) {
    int t  = wid + 8 * i;              // 32 tiles over 8 waves
    int km = t >> 3, sn = t & 7;
    int scol = sn * 16 + rlo;
    const _Float16* arow = &ca[(km * 16 + rlo) * CAS];  // A row (cluster), over c
    const _Float16* brow = &xnT[scol * XTS];            // B col (pixel), over c
    v8f acc = {};
    for (int cc4 = 0; cc4 < 4; ++cc4) {
      int kb = cc4 * 32;
      v16h af = frag2(arow + kb + hi * 8, arow + kb + 16 + hi * 8);
      v16h bf = frag2(brow + kb + hi * 8, brow + kb + 16 + hi * 8);
      acc = __builtin_amdgcn_wmma_f32_16x16x32_f16(false, af, false, bf,
                                                   (short)0, acc, false, false);
    }
    for (int r = 0; r < 8; ++r) {
      int k = km * 16 + hi * 8 + r;
      lg[k * LGS + scol] = 2.0f * ALPHA * acc[r] + biasl[k];
    }
  }
  __syncthreads();

  // ---- Phase 3: softmax over K per pixel; store probs f32 (lg) + f16 (ca) ----
  if (tid < TS) {
    int s = tid;
    float m = -1e30f;
    for (int k = 0; k < KK; ++k) m = fmaxf(m, lg[k * LGS + s]);
    float sum = 0.f;
    for (int k = 0; k < KK; ++k) {
      float e = __expf(lg[k * LGS + s] - m);
      lg[k * LGS + s] = e;
      sum += e;
    }
    float rs = 1.0f / sum;
    for (int k = 0; k < KK; ++k) {
      float a = lg[k * LGS + s] * rs;
      lg[k * LGS + s] = a;
      ca[k * CAS + s] = (_Float16)a;   // ca now holds 'a' (centroids dead)
    }
  }
  __syncthreads();

  // ---- Phase 3b: per-cluster assignment mass -> global asum ----
  if (tid < KK) {
    float rs = 0.f;
    for (int s = 0; s < TS; ++s) rs += lg[tid * LGS + s];
    atomicAdd(&asum[n * KK + tid], rs);
  }
  // EXEC back to all-ones before WMMA below

  // ---- Phase 4: vlad partial [64 x 128] = a[64 x TS] @ xhat^T[TS x 128] ----
  for (int i = 0; i < 4; ++i) {
    int t  = wid + 8 * i;
    int km = t >> 3, cn = t & 7;
    const _Float16* arow = &ca[(km * 16 + rlo) * CAS];        // A row (cluster), over s
    const _Float16* brow = &xn[(cn * 16 + rlo) * XNS];        // B col (channel), over s
    v8f acc = {};
    for (int ss4 = 0; ss4 < TS / 32; ++ss4) {
      int sb = ss4 * 32;
      v16h af = frag2(arow + sb + hi * 8, arow + sb + 16 + hi * 8);
      v16h bf = frag2(brow + sb + hi * 8, brow + sb + 16 + hi * 8);
      acc = __builtin_amdgcn_wmma_f32_16x16x32_f16(false, af, false, bf,
                                                   (short)0, acc, false, false);
    }
    for (int r = 0; r < 8; ++r) {
      int k = km * 16 + hi * 8 + r;
      int c = cn * 16 + rlo;
      atomicAdd(&vlad[((long)n * KK + k) * CC + c], acc[r]);  // 1MB L2-resident target
    }
  }
}

// Finalize: subtract asum*centroids, intra-norm over C, global norm over K*C
__global__ __launch_bounds__(256)
void netvlad_final(const float* __restrict__ vlad, const float* __restrict__ asum,
                   const float* __restrict__ cent, float* __restrict__ out) {
  __shared__ float v[KK * CC];         // 32 KB
  __shared__ float red[256];
  __shared__ float rnk[KK];
  __shared__ float rtot;
  const int n   = blockIdx.x;
  const int tid = threadIdx.x;
  const int k   = tid >> 2;            // 4 threads per cluster
  const int q   = tid & 3;

  const float ak = asum[n * KK + k];
  const float* vp = vlad + (long)n * KK * CC + k * CC;
  const float* cp = cent + k * CC;

  float ssq = 0.f;
  for (int j = 0; j < 32; ++j) {
    int c = q * 32 + j;
    float val = vp[c] - ak * cp[c];
    v[k * CC + c] = val;
    ssq += val * val;
  }
  red[tid] = ssq;
  __syncthreads();
  if (q == 0) {
    float s = red[tid] + red[tid + 1] + red[tid + 2] + red[tid + 3];
    rnk[k] = 1.0f / fmaxf(sqrtf(s), EPSN);
  }
  __syncthreads();

  float r1 = rnk[k];
  float ssq2 = 0.f;
  for (int j = 0; j < 32; ++j) {
    int c = q * 32 + j;
    float val = v[k * CC + c] * r1;
    v[k * CC + c] = val;
    ssq2 += val * val;
  }
  red[tid] = ssq2;
  __syncthreads();
  for (int off = 128; off > 0; off >>= 1) {
    if (tid < off) red[tid] += red[tid + off];
    __syncthreads();
  }
  if (tid == 0) rtot = 1.0f / fmaxf(sqrtf(red[0]), EPSN);
  __syncthreads();

  float r2 = rtot;
  float* op = out + (long)n * KK * CC;
  for (int j = 0; j < 32; ++j) {
    int c = q * 32 + j;
    op[k * CC + c] = v[k * CC + c] * r2;
  }
}

extern "C" void kernel_launch(void* const* d_in, const int* in_sizes, int n_in,
                              void* d_out, int out_size, void* d_ws, size_t ws_size,
                              hipStream_t stream) {
  const float* x    = (const float*)d_in[0];   // [32,128,64,64]
  const float* cent = (const float*)d_in[1];   // [64,128]
  float* vlad = (float*)d_ws;                  // [32,64,128] accumulator
  float* asum = vlad + (size_t)NN * KK * CC;   // [32,64]

  const int nz = NN * KK * CC + NN * KK;
  zero_ws_kernel<<<(nz + 255) / 256, 256, 0, stream>>>(vlad, nz);
  netvlad_main<<<NN * (SS / TS), 256, 0, stream>>>(x, cent, vlad, asum);
  netvlad_final<<<NN, 256, 0, stream>>>(vlad, asum, cent, (float*)d_out);
}
